// _StdAttention_73306501808594
// MI455X (gfx1250) — matrix operand order repaired and checked by reading, not compiled, not run on hardware
//
#include <hip/hip_runtime.h>

// ---------------------------------------------------------------------------
// GQA causal attention block for MI455X (gfx1250, wave32, WMMA bf16).
// B=2, S=2048, HIDDEN=1024, 16 Q heads / 4 KV heads, head_dim=64.
// All GEMM-shaped work goes through v_wmma_f32_16x16x32_bf16.
// ---------------------------------------------------------------------------

#define SEQ    2048
#define BATCH  2
#define HID    1024
#define NH     16
#define NKV    4
#define HD     64

typedef __bf16 bf16_t;
typedef __attribute__((ext_vector_type(16))) __bf16 v16bf;
typedef __attribute__((ext_vector_type(8)))  __bf16 v8bf;
typedef __attribute__((ext_vector_type(8)))  float  v8f;

__device__ __forceinline__ v8f wmma_bf16(v16bf a, v16bf b, v8f c) {
  // 8 args: (neg_a, A, neg_b, B, c_mod, C, reuse_a, reuse_b)
  return __builtin_amdgcn_wmma_f32_16x16x32_bf16(false, a, false, b, (short)0, c,
                                                 false, false);
}

// A-operand 16(M)x32(K) from row-major memory. rowptr = &A[row(lane)][k].
// lo lanes hold K 0..15 contiguous; hi lanes K 16..31.
__device__ __forceinline__ v16bf load_a(const bf16_t* rowptr, int lane) {
  const bf16_t* p = rowptr + ((lane & 16) ? 16 : 0);
  union { v16bf v; v8bf h[2]; } u;
  u.h[0] = *(const v8bf*)p;
  u.h[1] = *(const v8bf*)(p + 8);
  return u.v;
}

// B-operand 32(K)x16(N) from N-major memory (row n of source = column n of B).
// lo lanes hold K 0..15 contiguous; hi lanes K 16..31.
__device__ __forceinline__ v16bf load_b(const bf16_t* rowptr, int lane) {
  return *(const v16bf*)(rowptr + ((lane & 16) ? 16 : 0));
}

// ---------------------------------------------------------------------------
__global__ void cvt_f32_to_bf16_kernel(const float* __restrict__ in,
                                       bf16_t* __restrict__ out, int n) {
  int i = blockIdx.x * blockDim.x + threadIdx.x;
  if (i < n) out[i] = (bf16_t)in[i];
}

// ---------------------------------------------------------------------------
// C[M,N] = A[M,K] @ W[N,K]^T, bf16 in, f32 accumulate. One wave per 32x64 strip
// (8 WMMAs per 12 b128 loads in the K loop).
// mode 0: bf16 out * 1/8 (attn scale prefolded) -> [B,16,S,64]  (Q)
// mode 1: bf16 out -> [B,4,S,64]    (K)
// mode 2: bf16 out -> [B,4,64,S]    (V transposed)
// mode 3: f32 out row-major [M,N]   (final projection)
__global__ void __launch_bounds__(128) gemm_bf16_kernel(
    const bf16_t* __restrict__ A, const bf16_t* __restrict__ W,
    void* __restrict__ out, int M, int N, int K, int mode) {
  const int lane = threadIdx.x & 31;
  const int wid  = blockIdx.x * (blockDim.x >> 5) + (threadIdx.x >> 5);
  const int ntn  = N >> 6;
  if (wid >= (M >> 5) * ntn) return;   // wave-uniform: EXEC stays full for WMMA
  const int m0 = (wid / ntn) << 5;
  const int n0 = (wid % ntn) << 6;

  v8f acc[2][4] = {};
  const bf16_t* arow0 = A + (size_t)(m0 + (lane & 15)) * K;
  const bf16_t* arow1 = arow0 + (size_t)16 * K;
  const bf16_t* bp[4] = {
      W + (size_t)(n0 +  0 + (lane & 15)) * K,
      W + (size_t)(n0 + 16 + (lane & 15)) * K,
      W + (size_t)(n0 + 32 + (lane & 15)) * K,
      W + (size_t)(n0 + 48 + (lane & 15)) * K};

  for (int k = 0; k < K; k += 32) {
    v16bf a0 = load_a(arow0 + k, lane);
    v16bf a1 = load_a(arow1 + k, lane);
#pragma unroll
    for (int j = 0; j < 4; ++j) {
      v16bf b = load_b(bp[j] + k, lane);
      acc[0][j] = wmma_bf16(a0, b, acc[0][j]);
      acc[1][j] = wmma_bf16(a1, b, acc[1][j]);
    }
  }

  // C/D layout: vgpr r -> M = r (lo lanes) / 8+r (hi lanes); N = lane&15.
  const int roff = (lane & 16) ? 8 : 0;
  const int col0 = n0 + (lane & 15);
  const float postscale = (mode == 0) ? 0.125f : 1.0f;  // 1/sqrt(64) prefold
#pragma unroll
  for (int i = 0; i < 2; ++i) {
#pragma unroll
    for (int j = 0; j < 4; ++j) {
      const int c = col0 + j * 16;
#pragma unroll
      for (int r = 0; r < 8; ++r) {
        const int gr = m0 + i * 16 + roff + r;
        const float val = acc[i][j][r];
        if (mode == 3) {
          ((float*)out)[(size_t)gr * N + c] = val;
        } else {
          const int b = gr >> 11;       // / SEQ
          const int s = gr & (SEQ - 1);
          const int h = c >> 6;
          const int d = c & 63;
          size_t idx;
          if (mode == 0)      idx = (((size_t)(b * NH  + h)) * SEQ + s) * HD + d;
          else if (mode == 1) idx = (((size_t)(b * NKV + h)) * SEQ + s) * HD + d;
          else                idx = (((size_t)(b * NKV + h)) * HD + d) * SEQ + s;
          ((bf16_t*)out)[idx] = (bf16_t)(val * postscale);
        }
      }
    }
  }
}

// ---------------------------------------------------------------------------
// One 32-key chunk of transposed flash attention.
// S^T = K·Q^T (query == lane), then O^T += V^T·P^T.
template <bool MASKED>
__device__ __forceinline__ void attn_chunk(
    int k0, int lane, int qg, int koff,
    const bf16_t* __restrict__ Kbase, const bf16_t* __restrict__ Vbase,
    const v16bf& qb0, const v16bf& qb1, float& m, float& l,
    v8f& acc0, v8f& acc1, v8f& acc2, v8f& acc3) {
  // scores for keys [k0,k0+16) and [k0+16,k0+32)
  v8f st0 = {}, st1 = {};
  {
    const bf16_t* kr0 = Kbase + (size_t)(k0 + (lane & 15)) * HD;
    st0 = wmma_bf16(load_a(kr0, lane), qb0, st0);
    st0 = wmma_bf16(load_a(kr0 + 32, lane), qb1, st0);
    const bf16_t* kr1 = kr0 + 16 * HD;
    st1 = wmma_bf16(load_a(kr1, lane), qb0, st1);
    st1 = wmma_bf16(load_a(kr1 + 32, lane), qb1, st1);
  }

  float s0[8], s1[8];
  float mloc = -1e30f;
#pragma unroll
  for (int r = 0; r < 8; ++r) {
    if (MASKED) {
      s0[r] = (k0 + koff + r <= qg)      ? st0[r] : -1e30f;
      s1[r] = (k0 + 16 + koff + r <= qg) ? st1[r] : -1e30f;
    } else {
      s0[r] = st0[r];
      s1[r] = st1[r];
    }
    mloc = fmaxf(mloc, fmaxf(s0[r], s1[r]));
  }
  mloc = fmaxf(mloc, __shfl_xor(mloc, 16, 32));
  const float mnew = fmaxf(m, mloc);
  const float corr = __expf(m - mnew);
  m = mnew;

  // exp + repack P^T into the B-operand layout:
  //   lo lane needs k {0..7} (own st0) and {8..15} (partner's st0)
  //   hi lane needs k {16..23} (partner's st1) and {24..31} (own st1)
  float lloc = 0.0f;
  union { v16bf v; v8bf h8[2]; } pb;
#pragma unroll
  for (int r = 0; r < 8; ++r) {
    const float e0 = __expf(s0[r] - mnew);
    const float e1 = __expf(s1[r] - mnew);
    lloc += e0 + e1;
    const float sendv = (lane & 16) ? e0 : e1;
    const float recv  = __shfl_xor(sendv, 16, 32);
    pb.h8[0][r] = (bf16_t)((lane & 16) ? recv : e0);
    pb.h8[1][r] = (bf16_t)((lane & 16) ? e1 : recv);
  }
  l = l * corr + lloc + __shfl_xor(lloc, 16, 32);

#pragma unroll
  for (int r = 0; r < 8; ++r) {
    acc0[r] *= corr; acc1[r] *= corr; acc2[r] *= corr; acc3[r] *= corr;
  }

  // O^T += V^T(tile d) x P^T ; V^T is A-operand from Vt (row = d, ld = SEQ)
  const bf16_t* vrow = Vbase + (size_t)(lane & 15) * SEQ + k0;
  acc0 = wmma_bf16(load_a(vrow + (size_t) 0 * SEQ, lane), pb.v, acc0);
  acc1 = wmma_bf16(load_a(vrow + (size_t)16 * SEQ, lane), pb.v, acc1);
  acc2 = wmma_bf16(load_a(vrow + (size_t)32 * SEQ, lane), pb.v, acc2);
  acc3 = wmma_bf16(load_a(vrow + (size_t)48 * SEQ, lane), pb.v, acc3);
}

// One wave handles 16 queries for one (b,h). Q is prescaled by 1/sqrt(64).
__global__ void __launch_bounds__(128) flash_attn_kernel(
    const bf16_t* __restrict__ Qh,   // [B,NH,S,HD], prescaled
    const bf16_t* __restrict__ Kh,   // [B,NKV,S,HD]
    const bf16_t* __restrict__ Vt,   // [B,NKV,HD,S]
    bf16_t* __restrict__ attn) {     // [B*S, NH*HD]
  const int lane = threadIdx.x & 31;
  const int wid  = blockIdx.x * (blockDim.x >> 5) + (threadIdx.x >> 5);
  const int qi = wid & (SEQ / 16 - 1);
  const int h  = (wid >> 7) & (NH - 1);
  const int b  = wid >> 11;
  const int kvh = h >> 2;            // n_rep = 4
  const int q0 = qi * 16;

  const bf16_t* Qbase = Qh + ((size_t)(b * NH  + h))   * SEQ * HD;
  const bf16_t* Kbase = Kh + ((size_t)(b * NKV + kvh)) * SEQ * HD;
  const bf16_t* Vbase = Vt + ((size_t)(b * NKV + kvh)) * HD * SEQ;

  // Q as B-operand (column q = lane&15), split over d chunks [0,32) and [32,64)
  const bf16_t* qrow = Qbase + (size_t)(q0 + (lane & 15)) * HD;
  const v16bf qb0 = load_b(qrow +  0, lane);
  const v16bf qb1 = load_b(qrow + 32, lane);

  v8f acc0 = {}, acc1 = {}, acc2 = {}, acc3 = {};
  float m = -1e30f, l = 0.0f;
  const int qg   = q0 + (lane & 15);
  const int koff = (lane & 16) ? 8 : 0;
  const int nkeys = (qi + 1) * 16;                       // causal key range
  const int kend  = (q0 >= 31) ? ((((q0 - 31) >> 5) << 5) + 32) : 0;  // unmasked

  int k0 = 0;
  for (; k0 < kend; k0 += 32)
    attn_chunk<false>(k0, lane, qg, koff, Kbase, Vbase, qb0, qb1, m, l,
                      acc0, acc1, acc2, acc3);
  for (; k0 < nkeys; k0 += 32)   // 1-2 diagonal chunks with causal masking
    attn_chunk<true>(k0, lane, qg, koff, Kbase, Vbase, qb0, qb1, m, l,
                     acc0, acc1, acc2, acc3);

  // store O (normalized): C layout of O^T -> d = tile*16 + roff + r, q = lane&15
  const float invl = 1.0f / l;
  const int roff = (lane & 16) ? 8 : 0;
  const size_t rowbase = ((size_t)b * SEQ + q0 + (lane & 15)) * (NH * HD) + h * HD;
#pragma unroll
  for (int r = 0; r < 8; ++r) {
    attn[rowbase +  0 + roff + r] = (bf16_t)(acc0[r] * invl);
    attn[rowbase + 16 + roff + r] = (bf16_t)(acc1[r] * invl);
    attn[rowbase + 32 + roff + r] = (bf16_t)(acc2[r] * invl);
    attn[rowbase + 48 + roff + r] = (bf16_t)(acc3[r] * invl);
  }
}

// ---------------------------------------------------------------------------
extern "C" void kernel_launch(void* const* d_in, const int* in_sizes, int n_in,
                              void* d_out, int out_size, void* d_ws, size_t ws_size,
                              hipStream_t stream) {
  (void)in_sizes; (void)n_in; (void)out_size; (void)ws_size;
  const float* X  = (const float*)d_in[0];   // [2,2048,1024]
  const float* Wq = (const float*)d_in[1];   // [1024,1024]
  const float* Wk = (const float*)d_in[2];   // [256,1024]
  const float* Wv = (const float*)d_in[3];   // [256,1024]
  const float* Wo = (const float*)d_in[4];   // [1024,1024]
  float* out = (float*)d_out;

  const int M   = BATCH * SEQ;     // 4096
  const int nX  = M * HID;         // 4194304
  const int nWq = HID * HID;       // 1048576
  const int nWk = NKV * HD * HID;  // 262144
  const int nWv = nWk;
  const int nWo = nWq;

  char* ws = (char*)d_ws;
  size_t off = 0;
  bf16_t* Xb  = (bf16_t*)(ws + off); off += (size_t)nX  * 2;
  bf16_t* Wqb = (bf16_t*)(ws + off); off += (size_t)nWq * 2;
  bf16_t* Wkb = (bf16_t*)(ws + off); off += (size_t)nWk * 2;
  bf16_t* Wvb = (bf16_t*)(ws + off); off += (size_t)nWv * 2;
  bf16_t* Wob = (bf16_t*)(ws + off); off += (size_t)nWo * 2;
  bf16_t* Qh  = (bf16_t*)(ws + off); off += (size_t)BATCH * NH  * SEQ * HD * 2;
  bf16_t* Kh  = (bf16_t*)(ws + off); off += (size_t)BATCH * NKV * SEQ * HD * 2;
  bf16_t* Vt  = (bf16_t*)(ws + off); off += (size_t)BATCH * NKV * HD * SEQ * 2;
  bf16_t* att = (bf16_t*)(ws + off); off += (size_t)M * HID * 2;   // ~34.6 MB total

  // fp32 -> bf16 conversions
  cvt_f32_to_bf16_kernel<<<(nX  + 255) / 256, 256, 0, stream>>>(X,  Xb,  nX);
  cvt_f32_to_bf16_kernel<<<(nWq + 255) / 256, 256, 0, stream>>>(Wq, Wqb, nWq);
  cvt_f32_to_bf16_kernel<<<(nWk + 255) / 256, 256, 0, stream>>>(Wk, Wkb, nWk);
  cvt_f32_to_bf16_kernel<<<(nWv + 255) / 256, 256, 0, stream>>>(Wv, Wvb, nWv);
  cvt_f32_to_bf16_kernel<<<(nWo + 255) / 256, 256, 0, stream>>>(Wo, Wob, nWo);

  // QKV projections (waves = (M/32)*(N/64); 4 waves per 128-thread block)
  gemm_bf16_kernel<<<(M / 32) * (HID / 64) / 4, 128, 0, stream>>>(
      Xb, Wqb, Qh, M, HID, HID, 0);
  gemm_bf16_kernel<<<(M / 32) * (NKV * HD / 64) / 4, 128, 0, stream>>>(
      Xb, Wkb, Kh, M, NKV * HD, HID, 1);
  gemm_bf16_kernel<<<(M / 32) * (NKV * HD / 64) / 4, 128, 0, stream>>>(
      Xb, Wvb, Vt, M, NKV * HD, HID, 2);

  // attention: one wave per (b, h, 16-query tile) => 2*16*128 = 4096 waves
  flash_attn_kernel<<<BATCH * NH * (SEQ / 16) / 4, 128, 0, stream>>>(Qh, Kh, Vt, att);

  // output projection -> f32 d_out
  gemm_bf16_kernel<<<(M / 32) * (HID / 64) / 4, 128, 0, stream>>>(
      att, Wob, out, M, HID, HID, 3);
}
